// MineralDepositGAT_21517786153590
// MI455X (gfx1250) — matrix operand order repaired and checked by reading, not compiled, not run on hardware
//
#include <hip/hip_runtime.h>

// ---------------------------------------------------------------------------
// MineralDepositGAT for MI455X (gfx1250, wave32).
// GEMMs run on the matrix pipe via V_WMMA_F32_16X16X32_BF16 (bf16 operands,
// fp32 accumulation), 80x64 block tiles, 5 WMMAs per wave per K-step with
// B-fragment reuse.  Tiles staged to LDS with 128-bit loads/stores.
// Edge softmax/aggregation is the bandwidth-bound part (f32 global atomics).
// ---------------------------------------------------------------------------

#define THREADS 256
#define MT 80            // rows per block tile (50000 % 80 == 0)
#define NT 64            // cols per block tile (4 waves x 16)

typedef __attribute__((ext_vector_type(16))) __bf16 v16bf;
typedef __attribute__((ext_vector_type(8)))  float  v8f;

union FragAB { unsigned int u[8]; v16bf v; };

__device__ __forceinline__ unsigned short f2bf(float f) {
    unsigned int u = __float_as_uint(f);
    u += 0x7FFFu + ((u >> 16) & 1u);          // round-to-nearest-even
    return (unsigned short)(u >> 16);
}
// order-preserving float <-> uint encoding for atomic max
__device__ __forceinline__ unsigned int f2ord(float f) {
    unsigned int u = __float_as_uint(f);
    return (u & 0x80000000u) ? ~u : (u | 0x80000000u);
}
__device__ __forceinline__ float ord2f(unsigned int u) {
    unsigned int b = (u & 0x80000000u) ? (u ^ 0x80000000u) : ~u;
    return __uint_as_float(b);
}

// -------------------------- elementwise helpers ----------------------------

__global__ void k_fill_u32(unsigned int* p, unsigned int val, long count) {
    long i = (long)blockIdx.x * blockDim.x + threadIdx.x;
    if (i < count) p[i] = val;
}

__global__ void k_f32_to_bf16(const float* __restrict__ s,
                              unsigned short* __restrict__ d, long count) {
    long i = (long)blockIdx.x * blockDim.x + threadIdx.x;
    if (i < count) d[i] = f2bf(s[i]);
}

// pad weight matrix columns with zeros (fc3: 5 -> 16 cols)
__global__ void k_f32_to_bf16_pad(const float* __restrict__ s,
                                  unsigned short* __restrict__ d,
                                  int rows, int cols, int cols_pad) {
    int i = blockIdx.x * blockDim.x + threadIdx.x;
    if (i >= rows * cols_pad) return;
    int r = i / cols_pad, c = i - r * cols_pad;
    d[i] = (c < cols) ? f2bf(s[r * cols + c]) : (unsigned short)0;
}

// ------------------------------ WMMA GEMM ----------------------------------
// C[M,Np] = A[M,K] * B[K,Np]; A,B bf16 row-major, C f32.
// Block = 128 threads = 4 waves; block tile 80 rows x 64 cols.
// Each wave owns one 16-col slice (one B fragment per K-step, reused across
// 5 M-subtiles -> 5 WMMAs per wave per K-step).
// K multiple of 32, Np multiple of 16; all global tile accesses 16B-aligned.

__global__ __launch_bounds__(128)
void k_gemm_bf16_wmma(const unsigned short* __restrict__ A,
                      const unsigned short* __restrict__ B,
                      float* __restrict__ C, int M, int K, int Np) {
    __shared__ __align__(16) unsigned int   lds_a[MT * 16];     // 80 rows x 16 dwords
    __shared__ __align__(16) unsigned short lds_bt[NT * 32];    // B transposed [col][k]

    const int tid  = threadIdx.x;
    const int lane = tid & 31;
    const int wave = tid >> 5;
    const int m0 = blockIdx.x * MT;
    const int n0 = blockIdx.y * NT;
    const int nrem = Np - n0;              // > 0, multiple of 16
    const int rlow  = lane & 15;
    const int rhalf = lane >> 4;

    v8f acc[5];
#pragma unroll
    for (int s = 0; s < 5; ++s)
        acc[s] = (v8f){0.f, 0.f, 0.f, 0.f, 0.f, 0.f, 0.f, 0.f};

    for (int k0 = 0; k0 < K; k0 += 32) {
        // stage A tile: 80 rows x 32 bf16, 16B chunks (4 per row), row clamp
        // is branchless; 320 chunks / 128 threads.
#pragma unroll
        for (int i = tid; i < MT * 4; i += 128) {
            int row = i >> 2, q = i & 3;
            int rg = m0 + row;
            rg = (rg < M - 1) ? rg : (M - 1);
            const uint4 v = *reinterpret_cast<const uint4*>(
                                A + (size_t)rg * K + k0 + q * 8);
            *reinterpret_cast<uint4*>(&lds_a[row * 16 + q * 4]) = v;
        }
        // stage B tile transposed: 32 rows x 64 cols, 16B chunks (8 bf16),
        // scattered as 16-bit transposed DS stores.  Unconditional loads:
        // for every shape used here, indices past Np stay inside the Bbf
        // workspace carve and the affected waves never store their results.
#pragma unroll
        for (int i = tid; i < 32 * 8; i += 128) {
            int row = i >> 3, q = i & 7;
            const uint4 v = *reinterpret_cast<const uint4*>(
                                B + (size_t)(k0 + row) * Np + n0 + q * 8);
            const int cb = q * 8;
            lds_bt[(cb + 0) * 32 + row] = (unsigned short)(v.x & 0xFFFFu);
            lds_bt[(cb + 1) * 32 + row] = (unsigned short)(v.x >> 16);
            lds_bt[(cb + 2) * 32 + row] = (unsigned short)(v.y & 0xFFFFu);
            lds_bt[(cb + 3) * 32 + row] = (unsigned short)(v.y >> 16);
            lds_bt[(cb + 4) * 32 + row] = (unsigned short)(v.z & 0xFFFFu);
            lds_bt[(cb + 5) * 32 + row] = (unsigned short)(v.z >> 16);
            lds_bt[(cb + 6) * 32 + row] = (unsigned short)(v.w & 0xFFFFu);
            lds_bt[(cb + 7) * 32 + row] = (unsigned short)(v.w >> 16);
        }
        __syncthreads();

        // B fragment: V0-3 K=0..7 (+8 for lane>=16), V4-7 K=16..23 (+8)
        FragAB fb;
        const unsigned int* btd = reinterpret_cast<const unsigned int*>(lds_bt);
        const int bcol = wave * 16 + rlow;
#pragma unroll
        for (int v = 0; v < 8; ++v)
            fb.u[v] = btd[bcol * 16 + (v >> 2) * 8 + (v & 3) + rhalf * 4];

#pragma unroll
        for (int s = 0; s < 5; ++s) {
            // A fragment: V0-3 K=0..7 (+8 for lane>=16), V4-7 K=16..23 (+8)
            FragAB fa;
#pragma unroll
            for (int v = 0; v < 8; ++v)
                fa.u[v] = lds_a[(s * 16 + rlow) * 16 +
                                (v >> 2) * 8 + (v & 3) + rhalf * 4];
            acc[s] = __builtin_amdgcn_wmma_f32_16x16x32_bf16(
                         false, fa.v, false, fb.v, (short)0, acc[s], false, false);
        }
        __syncthreads();
    }

    if (wave * 16 < nrem) {
        int n = n0 + wave * 16 + rlow;
#pragma unroll
        for (int s = 0; s < 5; ++s) {
            int mb = m0 + s * 16 + rhalf * 8;
#pragma unroll
            for (int v = 0; v < 8; ++v)
                if (mb + v < M)
                    C[(size_t)(mb + v) * Np + n] = acc[s][v];
        }
    }
}

// --------------------------- GAT attention ---------------------------------

// per-node attention coefficients: a_src[i,h] = <xw[i,h,:], att_src[h,:]>
__global__ void k_attn_prep(const float* __restrict__ xw,
                            const float* __restrict__ att_s,
                            const float* __restrict__ att_d,
                            float* __restrict__ as_, float* __restrict__ ad_,
                            int n, int H, int CH) {
    int t = blockIdx.x * blockDim.x + threadIdx.x;
    if (t >= n * H) return;
    int i = t / H, h = t - i * H;
    const float* row = xw + (size_t)i * H * CH + h * CH;
    const float* ws = att_s + h * CH;
    const float* wd = att_d + h * CH;
    float ss = 0.f, sd = 0.f;
    for (int c = 0; c < CH; ++c) { float v = row[c]; ss += v * ws[c]; sd += v * wd[c]; }
    as_[t] = ss;
    ad_[t] = sd;
}

__device__ __forceinline__ void edge_sd(int e, int E, const int* se, const int* de,
                                        int& s, int& d) {
    if (e < E) { s = se[e]; d = de[e]; }       // data edges
    else       { s = e - E; d = e - E; }       // self loops
}

__global__ void k_edge_max(const int* __restrict__ se, const int* __restrict__ de,
                           const float* __restrict__ as_, const float* __restrict__ ad_,
                           unsigned int* __restrict__ smax, int E, int Etot, int H) {
    long t = (long)blockIdx.x * blockDim.x + threadIdx.x;
    if (t >= (long)Etot * H) return;
    int e = (int)(t / H), h = (int)(t - (long)e * H);
    int s, d; edge_sd(e, E, se, de, s, d);
    float v = as_[s * H + h] + ad_[d * H + h];
    v = (v > 0.f) ? v : 0.2f * v;                       // LeakyReLU(0.2)
    atomicMax(&smax[d * H + h], f2ord(v));
}

__global__ void k_edge_expsum(const int* __restrict__ se, const int* __restrict__ de,
                              const float* __restrict__ as_, const float* __restrict__ ad_,
                              const unsigned int* __restrict__ smax,
                              float* __restrict__ exb, float* __restrict__ den,
                              int E, int Etot, int H) {
    long t = (long)blockIdx.x * blockDim.x + threadIdx.x;
    if (t >= (long)Etot * H) return;
    int e = (int)(t / H), h = (int)(t - (long)e * H);
    int s, d; edge_sd(e, E, se, de, s, d);
    float v = as_[s * H + h] + ad_[d * H + h];
    v = (v > 0.f) ? v : 0.2f * v;
    float m  = ord2f(smax[d * H + h]);
    float ex = __expf(v - m);
    exb[t] = ex;
    atomicAdd(&den[d * H + h], ex);
}

// out[dst, :] += alpha * xw[src, :]; thread = (edge, 4-channel group)
__global__ void k_edge_aggr(const int* __restrict__ se, const int* __restrict__ de,
                            const float* __restrict__ exb, const float* __restrict__ den,
                            const float* __restrict__ xw, float* __restrict__ agg,
                            int E, int Etot, int H, int HC) {
    int G = HC >> 2;
    long t = (long)blockIdx.x * blockDim.x + threadIdx.x;
    if (t >= (long)Etot * G) return;
    int e = (int)(t / G), j = (int)(t - (long)e * G);
    int c0 = j * 4, h = c0 >> 6;                        // CH = 64
    int s, d; edge_sd(e, E, se, de, s, d);
    const float* src_row = xw + (size_t)s * HC + c0;
    __builtin_prefetch(src_row, 0, 1);                  // global_prefetch_b8
    float alpha = exb[(long)e * H + h] / (den[d * H + h] + 1e-16f);
    const float4 v = *reinterpret_cast<const float4*>(src_row);
    float* o = agg + (size_t)d * HC + c0;
    atomicAdd(o + 0, alpha * v.x);
    atomicAdd(o + 1, alpha * v.y);
    atomicAdd(o + 2, alpha * v.z);
    atomicAdd(o + 3, alpha * v.w);
}

// ------------------------- post-processing ---------------------------------

__global__ void k_bias_elu(const float* __restrict__ agg, const float* __restrict__ bias,
                           float* __restrict__ act, int n, int C) {
    long t = (long)blockIdx.x * blockDim.x + threadIdx.x;
    if (t >= (long)n * C) return;
    int c = (int)(t % C);
    float v = agg[t] + bias[c];
    act[t] = (v > 0.f) ? v : (__expf(v) - 1.f);          // ELU
}

// one block per column: mean and rstd over n rows
__global__ void k_col_stats(const float* __restrict__ act, float* __restrict__ stats,
                            int n, int C) {
    __shared__ float s1[256], s2[256];
    int c = blockIdx.x, tid = threadIdx.x;
    float a = 0.f, b = 0.f;
    for (int r = tid; r < n; r += 256) {
        float v = act[(size_t)r * C + c];
        a += v; b += v * v;
    }
    s1[tid] = a; s2[tid] = b;
    __syncthreads();
    for (int s = 128; s > 0; s >>= 1) {
        if (tid < s) { s1[tid] += s1[tid + s]; s2[tid] += s2[tid + s]; }
        __syncthreads();
    }
    if (tid == 0) {
        float mean = s1[0] / n;
        float var  = s2[0] / n - mean * mean;
        stats[c]     = mean;
        stats[C + c] = rsqrtf(var + 1e-5f);
    }
}

// BN apply, write bf16 activation for the next WMMA GEMM
__global__ void k_bn_apply(const float* __restrict__ act, const float* __restrict__ stats,
                           const float* __restrict__ g, const float* __restrict__ b,
                           unsigned short* __restrict__ out_bf, int n, int C) {
    long t = (long)blockIdx.x * blockDim.x + threadIdx.x;
    if (t >= (long)n * C) return;
    int c = (int)(t % C);
    float v = (act[t] - stats[c]) * stats[C + c] * g[c] + b[c];
    out_bf[t] = f2bf(v);
}

// bias + tanh-approx GELU (jax.nn.gelu default), write bf16 for next GEMM
__global__ void k_bias_gelu(const float* __restrict__ Cin, const float* __restrict__ bias,
                            unsigned short* __restrict__ out_bf, int n, int C) {
    long t = (long)blockIdx.x * blockDim.x + threadIdx.x;
    if (t >= (long)n * C) return;
    int c = (int)(t % C);
    float x = Cin[t] + bias[c];
    float u = 0.7978845608028654f * (x + 0.044715f * x * x * x);
    float y = 0.5f * x * (1.f + tanhf(u));
    out_bf[t] = f2bf(y);
}

// final: unpad 16 -> 5 cols and add fc3 bias
__global__ void k_bias_out(const float* __restrict__ Cin, const float* __restrict__ bias,
                           float* __restrict__ out, int n) {
    long t = (long)blockIdx.x * blockDim.x + threadIdx.x;
    if (t >= (long)n * 5) return;
    int i = (int)(t / 5), q = (int)(t - (long)i * 5);
    out[t] = Cin[(size_t)i * 16 + q] + bias[q];
}

// ------------------------------ driver -------------------------------------
// d_in order (setup_inputs dict insertion order, nested dicts flattened in
// insertion order):
//  0 x  1 edge_index  2 edge_attr(unused)
//  3 conv0.W  4 conv0.att_src  5 conv0.att_dst  6 conv0.b
//  7 conv1.W  8 conv1.att_src  9 conv1.att_dst 10 conv1.b
// 11 conv2.W 12 conv2.att_src 13 conv2.att_dst 14 conv2.b
// 15 bn0_g 16 bn0_b 17 bn1_g 18 bn1_b 19 bn2_g 20 bn2_b
// 21 fc1_W 22 fc1_b 23 fc2_W 24 fc2_b 25 fc3_W 26 fc3_b

static inline int cdivl(long a, long b) { return (int)((a + b - 1) / b); }

extern "C" void kernel_launch(void* const* d_in, const int* in_sizes, int n_in,
                              void* d_out, int out_size, void* d_ws, size_t ws_size,
                              hipStream_t stream) {
    const float* x    = (const float*)d_in[0];
    const int*   eidx = (const int*)d_in[1];
    const int n = in_sizes[0] / 32;            // 50000
    const int E = in_sizes[1] / 2;             // 800000
    const int Etot = E + n;                    // + self loops
    const int* esrc = eidx;
    const int* edst = eidx + E;

    const float* W[3]   = {(const float*)d_in[3], (const float*)d_in[7],  (const float*)d_in[11]};
    const float* atS[3] = {(const float*)d_in[4], (const float*)d_in[8],  (const float*)d_in[12]};
    const float* atD[3] = {(const float*)d_in[5], (const float*)d_in[9],  (const float*)d_in[13]};
    const float* cb[3]  = {(const float*)d_in[6], (const float*)d_in[10], (const float*)d_in[14]};
    const float* bng[3] = {(const float*)d_in[15], (const float*)d_in[17], (const float*)d_in[19]};
    const float* bnb[3] = {(const float*)d_in[16], (const float*)d_in[18], (const float*)d_in[20]};
    const int Kl[3] = {32, 320, 320};
    const int Hl[3] = {5, 5, 1};
    const int CH = 64;

    // ---- workspace carve-up (~246 MB) ----
    char* base = (char*)d_ws;
    size_t off = 0;
    auto carve = [&](size_t bytes) -> void* {
        void* p = base + off;
        off = (off + bytes + 255) & ~(size_t)255;
        return p;
    };
    unsigned short* Abf  = (unsigned short*)carve((size_t)n * 320 * 2);
    unsigned short* Bbf  = (unsigned short*)carve((size_t)320 * 320 * 2);
    float*          Cf   = (float*)carve((size_t)n * 320 * 4);
    float*          agg  = (float*)carve((size_t)n * 320 * 4);
    float*          act  = (float*)carve((size_t)n * 320 * 4);
    float*          asrc = (float*)carve((size_t)n * 5 * 4);
    float*          adst = (float*)carve((size_t)n * 5 * 4);
    unsigned int*   smax = (unsigned int*)carve((size_t)n * 5 * 4);
    float*          den  = (float*)carve((size_t)n * 5 * 4);
    float*          exb  = (float*)carve((size_t)Etot * 5 * 4);
    float*          stats= (float*)carve(2 * 320 * 4);
    (void)ws_size; (void)n_in; (void)out_size;

    // layer-0 A operand = bf16(x)
    k_f32_to_bf16<<<cdivl((long)n * 32, THREADS), THREADS, 0, stream>>>(x, Abf, (long)n * 32);

    // ---- three GAT conv layers ----
    for (int L = 0; L < 3; ++L) {
        const int K = Kl[L], H = Hl[L], HC = H * CH;

        k_f32_to_bf16<<<cdivl((long)K * HC, THREADS), THREADS, 0, stream>>>(W[L], Bbf, (long)K * HC);

        dim3 gg(cdivl(n, MT), cdivl(HC, NT));
        k_gemm_bf16_wmma<<<gg, 128, 0, stream>>>(Abf, Bbf, Cf, n, K, HC);

        k_attn_prep<<<cdivl((long)n * H, THREADS), THREADS, 0, stream>>>(
            Cf, atS[L], atD[L], asrc, adst, n, H, CH);

        k_fill_u32<<<cdivl((long)n * H, THREADS), THREADS, 0, stream>>>(smax, 0x007FFFFFu, (long)n * H); // ord(-inf)
        k_fill_u32<<<cdivl((long)n * H, THREADS), THREADS, 0, stream>>>((unsigned int*)den, 0u, (long)n * H);
        k_fill_u32<<<cdivl((long)n * HC, THREADS), THREADS, 0, stream>>>((unsigned int*)agg, 0u, (long)n * HC);

        k_edge_max<<<cdivl((long)Etot * H, THREADS), THREADS, 0, stream>>>(
            esrc, edst, asrc, adst, smax, E, Etot, H);
        k_edge_expsum<<<cdivl((long)Etot * H, THREADS), THREADS, 0, stream>>>(
            esrc, edst, asrc, adst, smax, exb, den, E, Etot, H);
        k_edge_aggr<<<cdivl((long)Etot * (HC / 4), THREADS), THREADS, 0, stream>>>(
            esrc, edst, exb, den, Cf, agg, E, Etot, H, HC);

        k_bias_elu<<<cdivl((long)n * HC, THREADS), THREADS, 0, stream>>>(agg, cb[L], act, n, HC);
        k_col_stats<<<HC, 256, 0, stream>>>(act, stats, n, HC);
        k_bn_apply<<<cdivl((long)n * HC, THREADS), THREADS, 0, stream>>>(
            act, stats, bng[L], bnb[L], Abf, n, HC);
    }

    // ---- MLP head ----
    // fc1: [n,64] @ [64,128] + gelu
    k_f32_to_bf16<<<cdivl(64L * 128, THREADS), THREADS, 0, stream>>>((const float*)d_in[21], Bbf, 64L * 128);
    k_gemm_bf16_wmma<<<dim3(cdivl(n, MT), 2), 128, 0, stream>>>(Abf, Bbf, Cf, n, 64, 128);
    k_bias_gelu<<<cdivl((long)n * 128, THREADS), THREADS, 0, stream>>>(Cf, (const float*)d_in[22], Abf, n, 128);
    // fc2: [n,128] @ [128,64] + gelu
    k_f32_to_bf16<<<cdivl(128L * 64, THREADS), THREADS, 0, stream>>>((const float*)d_in[23], Bbf, 128L * 64);
    k_gemm_bf16_wmma<<<dim3(cdivl(n, MT), 1), 128, 0, stream>>>(Abf, Bbf, Cf, n, 128, 64);
    k_bias_gelu<<<cdivl((long)n * 64, THREADS), THREADS, 0, stream>>>(Cf, (const float*)d_in[24], Abf, n, 64);
    // fc3: [n,64] @ [64,5] (cols padded to 16)
    k_f32_to_bf16_pad<<<cdivl(64L * 16, THREADS), THREADS, 0, stream>>>((const float*)d_in[25], Bbf, 64, 5, 16);
    k_gemm_bf16_wmma<<<dim3(cdivl(n, MT), 1), 128, 0, stream>>>(Abf, Bbf, Cf, n, 64, 16);
    k_bias_out<<<cdivl((long)n * 5, THREADS), THREADS, 0, stream>>>(Cf, (const float*)d_in[26], (float*)d_out, n);
}
